// DecoderBlock_27771258536816
// MI455X (gfx1250) — matrix operand order repaired and checked by reading, not compiled, not run on hardware
//
#include <hip/hip_runtime.h>
#include <cstddef>
#include <cstdint>

#define B_   4
#define S_   1024
#define E_   1024
#define D_   1024
#define H_   16
#define DH_  64
#define DFF_ 4096
#define MTOK 4096   // B_*S_ == B_*E_

typedef __attribute__((ext_vector_type(16))) __bf16 v16bf;
typedef __attribute__((ext_vector_type(8)))  float  v8f;
typedef __attribute__((ext_vector_type(4)))  unsigned int v4u;
typedef __attribute__((ext_vector_type(8)))  int  v8i;
typedef __attribute__((ext_vector_type(4)))  int  v4i;

union Frag {
  v16bf  v;
  __bf16 e[16];
  uint4  q[2];
};

__device__ __forceinline__ v8f wmma_bf16(const Frag& a, const Frag& b, v8f c) {
  return __builtin_amdgcn_wmma_f32_16x16x32_bf16(false, a.v, false, b.v,
                                                 (short)0, c, false, false);
}

// ---------------------------------------------------------------------------
// TDM: 2-D tile load (bf16), global [rows][K-contig] -> LDS with 16B pad per
// 64B row (pad_interval = 16 DWORDs, pad_amount = 4 DWORDs) => LDS row stride
// of 40 bf16 elements, matching the WMMA fragment addressing below.
// ---------------------------------------------------------------------------
__device__ __forceinline__ void tdm_load_2d(unsigned lds_byte_off,
                                            const __bf16* gptr,
                                            unsigned tensor_d0, unsigned tensor_d1,
                                            unsigned long stride0_elems,
                                            unsigned tile_d0, unsigned tile_d1) {
  const unsigned long ga = (unsigned long)(uintptr_t)gptr;
  v4u g0;
  g0[0] = 1u;                                             // count=1 (valid user D#)
  g0[1] = lds_byte_off;                                   // lds_addr
  g0[2] = (unsigned)(ga & 0xFFFFFFFFu);                   // global_addr[31:0]
  g0[3] = (unsigned)((ga >> 32) & 0x1FFFFFFu) | (2u << 30); // addr[56:32], type=2
  v8i g1;
  // data_size=1 (2B), pad_enable, pad_interval=3 (16 DW), pad_amount=3 (4 DW)
  g1[0] = (int)((1u << 16) | (1u << 20) | (3u << 22) | (3u << 25));
  g1[1] = (int)((tensor_d0 & 0xFFFFu) << 16);             // dim0[15:0]
  g1[2] = (int)(((tensor_d0 >> 16) & 0xFFFFu) | ((tensor_d1 & 0xFFFFu) << 16));
  g1[3] = (int)(((tensor_d1 >> 16) & 0xFFFFu) | ((tile_d0 & 0xFFFFu) << 16));
  g1[4] = (int)(tile_d1 & 0xFFFFu);                       // tile_dim2 = 0
  g1[5] = (int)(stride0_elems & 0xFFFFFFFFu);
  g1[6] = (int)((stride0_elems >> 32) & 0xFFFFu);         // dim1_stride = 0 (2-D)
  g1[7] = 0;
  const v4i z4 = {0, 0, 0, 0};
  const v8i z8 = {0, 0, 0, 0, 0, 0, 0, 0};
  __builtin_amdgcn_tensor_load_to_lds(g0, g1, z4, z4, z8, 0);
}

__device__ __forceinline__ unsigned lds_off(const void* p) {
  return (unsigned)(uintptr_t)p;  // flat shared-aperture addr[31:0] == LDS offset
}

// ---------------------------------------------------------------------------
// GEMM: C[M,N] = A[M,K] @ Wt[N,K]^T + bias, A/Wt pre-packed bf16, K-contig.
// 256 threads = 8 waves (2x4); WG tile 64x256; wave tile 32x64; K-step 32.
// Double-buffered TDM staging (wave 0 issues, TENSORcnt pipelined).
// M%64==0, N%256==0, K%32==0 for all calls here.
// ---------------------------------------------------------------------------
template <int GELU>
__global__ __launch_bounds__(256) void gemm_tdm_kernel(
    const __bf16* __restrict__ A, const __bf16* __restrict__ Bt,
    const float* __restrict__ bias, float* __restrict__ C,
    int M, int N, int K) {
  __shared__ __align__(16) __bf16 lA[2][64 * 40];
  __shared__ __align__(16) __bf16 lB[2][256 * 40];

  const int tid  = threadIdx.x;
  const int wave = tid >> 5, lane = tid & 31;
  const int hf   = lane >> 4, l16 = lane & 15;
  const int wm   = wave & 1,  wn  = wave >> 1;   // 2 x 4 wave grid
  const int m0   = blockIdx.y * 64;
  const int n0   = blockIdx.x * 256;
  const bool w0  = (wave == 0);

  v8f acc[2][4] = {};
  const int nk = K >> 5;

  if (w0) {
    tdm_load_2d(lds_off(&lA[0][0]), A  + (size_t)m0 * K, K, M, (unsigned long)K, 32, 64);
    tdm_load_2d(lds_off(&lB[0][0]), Bt + (size_t)n0 * K, K, N, (unsigned long)K, 32, 256);
  }

  for (int kb = 0; kb < nk; ++kb) {
    if (w0) __builtin_amdgcn_s_wait_tensorcnt(0);
    __syncthreads();   // buf[kb&1] ready; all waves done with buf[(kb+1)&1]
    if (w0 && (kb + 1) < nk) {
      const int k1 = (kb + 1) << 5;
      tdm_load_2d(lds_off(&lA[(kb + 1) & 1][0]), A  + (size_t)m0 * K + k1,
                  K, M, (unsigned long)K, 32, 64);
      tdm_load_2d(lds_off(&lB[(kb + 1) & 1][0]), Bt + (size_t)n0 * K + k1,
                  K, N, (unsigned long)K, 32, 256);
    }
    const __bf16* cA = &lA[kb & 1][0];
    const __bf16* cB = &lB[kb & 1][0];

    Frag fa[2], fb[4];
    for (int mi = 0; mi < 2; ++mi) {
      // A layout: lane holds row m=l16; K chunks at hf*8 and 16+hf*8
      const __bf16* p = cA + (wm * 32 + mi * 16 + l16) * 40 + hf * 8;
      fa[mi].q[0] = *(const uint4*)p;
      fa[mi].q[1] = *(const uint4*)(p + 16);
    }
    for (int ni = 0; ni < 4; ++ni) {
      // B layout: lane holds column n=l16; K chunks at hf*8 and 16+hf*8
      const __bf16* p = cB + (wn * 64 + ni * 16 + l16) * 40 + hf * 8;
      fb[ni].q[0] = *(const uint4*)p;
      fb[ni].q[1] = *(const uint4*)(p + 16);
    }
    for (int mi = 0; mi < 2; ++mi)
      for (int ni = 0; ni < 4; ++ni)
        acc[mi][ni] = wmma_bf16(fa[mi], fb[ni], acc[mi][ni]);
    __syncthreads();
  }

  // epilogue: C row = hf*8 + r, col = l16
  for (int mi = 0; mi < 2; ++mi)
    for (int ni = 0; ni < 4; ++ni) {
      const int col = n0 + wn * 64 + ni * 16 + l16;
      const float bv = bias[col];
      for (int r = 0; r < 8; ++r) {
        const int row = m0 + wm * 32 + mi * 16 + hf * 8 + r;
        float v = acc[mi][ni][r] + bv;
        if (GELU) v = 0.5f * v * (1.0f + erff(v * 0.70710678118654752f));
        C[(size_t)row * N + col] = v;
      }
    }
}

// ---------------------------------------------------------------------------
// f32 -> bf16 elementwise pack (n % 1024 == 0)
// ---------------------------------------------------------------------------
__global__ __launch_bounds__(256) void cvt_bf16_kernel(
    const float* __restrict__ in, __bf16* __restrict__ out) {
  const size_t i = ((size_t)blockIdx.x * 256 + threadIdx.x) * 4;
  const float4 v = *(const float4*)(in + i);
  __bf16 t[4];
  t[0] = (__bf16)v.x; t[1] = (__bf16)v.y; t[2] = (__bf16)v.z; t[3] = (__bf16)v.w;
  *(uint2*)(out + i) = *(const uint2*)t;
}

// ---------------------------------------------------------------------------
// f32 [Kd][Nd] -> bf16 [Nd][Kd] transpose pack (Kd,Nd % 32 == 0)
// ---------------------------------------------------------------------------
__global__ __launch_bounds__(256) void transpose_bf16_kernel(
    const float* __restrict__ W, __bf16* __restrict__ Wt, int Kd, int Nd) {
  __shared__ float t[32][33];
  const int tid = threadIdx.x;
  const int k0 = blockIdx.y * 32, n0 = blockIdx.x * 32;
  const int c = tid & 31, r = tid >> 5;   // 8 rows per pass
  for (int i = 0; i < 4; ++i)
    t[r + i * 8][c] = W[(size_t)(k0 + r + i * 8) * Nd + n0 + c];
  __syncthreads();
  for (int i = 0; i < 4; ++i)
    Wt[(size_t)(n0 + r + i * 8) * Kd + k0 + c] = (__bf16)t[c][r + i * 8];
}

// ---------------------------------------------------------------------------
// Flash attention: WG = (q-block 64, head, batch), 128 threads = 4 waves,
// wave owns 16 q rows x DH=64; 32-key blocks, online softmax, bf16 WMMA.
// ---------------------------------------------------------------------------
__global__ __launch_bounds__(128) void attn_kernel(
    const float* __restrict__ Qb, const float* __restrict__ Kb,
    const float* __restrict__ Vb, float* __restrict__ Ob,
    long qBS, int qSS, int qHS,
    long kBS, int kSS, int kHS,
    long vBS, int vSS, int vHS,
    int kvlen, int causal) {
  __shared__ __align__(16) __bf16 lK[32 * 72];       // [key][dh]
  __shared__ __align__(16) __bf16 lV[64 * 40];       // [dh][key]
  __shared__ __align__(16) __bf16 lP[4 * 16 * 40];   // per-wave P tile

  const int tid  = threadIdx.x;
  const int wave = tid >> 5, lane = tid & 31;
  const int hf   = lane >> 4, l16 = lane & 15;
  const int b    = blockIdx.z, h = blockIdx.y;
  const int q0   = blockIdx.x * 64 + wave * 16;

  Frag aQ[2];
  {
    const float* qrow = Qb + (size_t)b * qBS + (size_t)h * qHS
                           + (size_t)(q0 + l16) * qSS;
    for (int ks = 0; ks < 2; ++ks) {
      const float* p0 = qrow + ks * 32 + hf * 8;
      float4 a0 = *(const float4*)p0;
      float4 a1 = *(const float4*)(p0 + 4);
      float4 b0 = *(const float4*)(p0 + 16);
      float4 b1 = *(const float4*)(p0 + 20);
      aQ[ks].e[0]  = (__bf16)a0.x; aQ[ks].e[1]  = (__bf16)a0.y;
      aQ[ks].e[2]  = (__bf16)a0.z; aQ[ks].e[3]  = (__bf16)a0.w;
      aQ[ks].e[4]  = (__bf16)a1.x; aQ[ks].e[5]  = (__bf16)a1.y;
      aQ[ks].e[6]  = (__bf16)a1.z; aQ[ks].e[7]  = (__bf16)a1.w;
      aQ[ks].e[8]  = (__bf16)b0.x; aQ[ks].e[9]  = (__bf16)b0.y;
      aQ[ks].e[10] = (__bf16)b0.z; aQ[ks].e[11] = (__bf16)b0.w;
      aQ[ks].e[12] = (__bf16)b1.x; aQ[ks].e[13] = (__bf16)b1.y;
      aQ[ks].e[14] = (__bf16)b1.z; aQ[ks].e[15] = (__bf16)b1.w;
    }
  }

  __bf16* lPw = &lP[wave * (16 * 40)];
  v8f oacc[4] = {};
  float m_run[8], l_run[8];
  for (int r = 0; r < 8; ++r) { m_run[r] = -1e30f; l_run[r] = 0.f; }

  const int nkb = causal ? ((blockIdx.x * 64 + 63) / 32 + 1) : (kvlen / 32);

  for (int kb = 0; kb < nkb; ++kb) {
    const int kk0 = kb * 32;
    __syncthreads();
    {
      const int krow = tid >> 4;        // 0..7
      const int d4   = (tid & 15) * 4;  // 0..60
      for (int i = 0; i < 4; ++i) {
        const int key = krow + i * 8;
        const float4 kv = *(const float4*)(Kb + (size_t)b * kBS
            + (size_t)h * kHS + (size_t)(kk0 + key) * kSS + d4);
        __bf16* kd = &lK[key * 72 + d4];
        kd[0] = (__bf16)kv.x; kd[1] = (__bf16)kv.y;
        kd[2] = (__bf16)kv.z; kd[3] = (__bf16)kv.w;
        const float4 vv = *(const float4*)(Vb + (size_t)b * vBS
            + (size_t)h * vHS + (size_t)(kk0 + key) * vSS + d4);
        lV[(d4 + 0) * 40 + key] = (__bf16)vv.x;
        lV[(d4 + 1) * 40 + key] = (__bf16)vv.y;
        lV[(d4 + 2) * 40 + key] = (__bf16)vv.z;
        lV[(d4 + 3) * 40 + key] = (__bf16)vv.w;
      }
    }
    __syncthreads();

    v8f sc[2] = {};
    for (int ks = 0; ks < 2; ++ks)
      for (int nt = 0; nt < 2; ++nt) {
        Frag bK;  // lane = key column; K(dh) chunks at ks*32 + hf*8 and ks*32 + 16 + hf*8
        const __bf16* p = &lK[(nt * 16 + l16) * 72 + ks * 32 + hf * 8];
        bK.q[0] = *(const uint4*)p;
        bK.q[1] = *(const uint4*)(p + 16);
        sc[nt] = wmma_bf16(aQ[ks], bK, sc[nt]);
      }

    float mloc[8];
    for (int r = 0; r < 8; ++r) {
      float v0 = sc[0][r] * 0.125f;     // 1/sqrt(DH)
      float v1 = sc[1][r] * 0.125f;
      if (causal) {
        const int qi = q0 + hf * 8 + r;
        if (kk0 + l16 > qi)      v0 = -1e30f;
        if (kk0 + 16 + l16 > qi) v1 = -1e30f;
      }
      sc[0][r] = v0; sc[1][r] = v1;
      mloc[r] = fmaxf(v0, v1);
    }
    for (int off = 1; off < 16; off <<= 1)
      for (int r = 0; r < 8; ++r)
        mloc[r] = fmaxf(mloc[r], __shfl_xor(mloc[r], off, 32));

    float rsum[8];
    for (int r = 0; r < 8; ++r) {
      const float mn = fmaxf(m_run[r], mloc[r]);
      const float alpha = __expf(m_run[r] - mn);
      m_run[r] = mn;
      const float p0 = __expf(sc[0][r] - mn);
      const float p1 = __expf(sc[1][r] - mn);
      rsum[r] = p0 + p1;
      l_run[r] *= alpha;
      for (int ni = 0; ni < 4; ++ni) oacc[ni][r] *= alpha;
      const int rr = hf * 8 + r;
      lPw[rr * 40 + l16]      = (__bf16)p0;
      lPw[rr * 40 + 16 + l16] = (__bf16)p1;
    }
    for (int off = 1; off < 16; off <<= 1)
      for (int r = 0; r < 8; ++r)
        rsum[r] += __shfl_xor(rsum[r], off, 32);
    for (int r = 0; r < 8; ++r) l_run[r] += rsum[r];

    Frag aP;  // P (16x32) as A operand; LDS is in-order within a wave
    {
      const __bf16* p = &lPw[l16 * 40 + hf * 8];
      aP.q[0] = *(const uint4*)p;
      aP.q[1] = *(const uint4*)(p + 16);
    }
    for (int ni = 0; ni < 4; ++ni) {
      Frag bV;  // lane = dh column; K(key) chunks at hf*8 and 16 + hf*8
      const __bf16* p = &lV[(ni * 16 + l16) * 40 + hf * 8];
      bV.q[0] = *(const uint4*)p;
      bV.q[1] = *(const uint4*)(p + 16);
      oacc[ni] = wmma_bf16(aP, bV, oacc[ni]);
    }
  }

  for (int ni = 0; ni < 4; ++ni)
    for (int r = 0; r < 8; ++r) {
      const int qi = q0 + hf * 8 + r;
      Ob[(size_t)b * ((size_t)S_ * D_) + (size_t)qi * D_
         + h * DH_ + ni * 16 + l16] = oacc[ni][r] / l_run[r];
    }
}

// ---------------------------------------------------------------------------
// y = LayerNorm(x + res) * g + b, rows of D_=1024
// ---------------------------------------------------------------------------
__global__ __launch_bounds__(256) void add_ln_kernel(
    const float* __restrict__ x, const float* __restrict__ res,
    const float* __restrict__ g, const float* __restrict__ be,
    float* __restrict__ y) {
  const int N = D_;
  const int row = blockIdx.x;
  const int tid = threadIdx.x;
  const float* xp = x + (size_t)row * N;
  const float* rp = res + (size_t)row * N;
  float vals[4], s = 0.f, s2 = 0.f;
  for (int i = 0; i < 4; ++i) {
    const int c = tid + i * 256;
    const float a = xp[c] + rp[c];
    vals[i] = a; s += a; s2 += a * a;
  }
  for (int off = 16; off > 0; off >>= 1) {
    s  += __shfl_xor(s, off, 32);
    s2 += __shfl_xor(s2, off, 32);
  }
  __shared__ float rsm[8], rs2[8], stats[2];
  const int wave = tid >> 5, lane = tid & 31;
  if (lane == 0) { rsm[wave] = s; rs2[wave] = s2; }
  __syncthreads();
  if (tid == 0) {
    float t = 0.f, t2 = 0.f;
    for (int i = 0; i < 8; ++i) { t += rsm[i]; t2 += rs2[i]; }
    const float mean = t / (float)N;
    const float var  = t2 / (float)N - mean * mean;
    stats[0] = mean;
    stats[1] = rsqrtf(var + 1e-5f);
  }
  __syncthreads();
  const float mean = stats[0], rstd = stats[1];
  float* yp = y + (size_t)row * N;
  for (int i = 0; i < 4; ++i) {
    const int c = tid + i * 256;
    yp[c] = g[c] * (vals[i] - mean) * rstd + be[c];
  }
}

// ---------------------------------------------------------------------------
extern "C" void kernel_launch(void* const* d_in, const int* in_sizes, int n_in,
                              void* d_out, int out_size, void* d_ws,
                              size_t ws_size, hipStream_t stream) {
  (void)in_sizes; (void)n_in; (void)out_size; (void)ws_size;
  const float* x      = (const float*)d_in[0];
  const float* enc_x  = (const float*)d_in[1];
  const float* w_qkv  = (const float*)d_in[2];
  const float* b_qkv  = (const float*)d_in[3];
  const float* w_sa_o = (const float*)d_in[4];
  const float* b_sa_o = (const float*)d_in[5];
  const float* w_q    = (const float*)d_in[6];
  const float* b_q    = (const float*)d_in[7];
  const float* w_k    = (const float*)d_in[8];
  const float* b_k    = (const float*)d_in[9];
  const float* w_v    = (const float*)d_in[10];
  const float* b_v    = (const float*)d_in[11];
  const float* w_ca_o = (const float*)d_in[12];
  const float* b_ca_o = (const float*)d_in[13];
  const float* w1     = (const float*)d_in[14];
  const float* b1     = (const float*)d_in[15];
  const float* w2     = (const float*)d_in[16];
  const float* b2     = (const float*)d_in[17];
  const float* g1     = (const float*)d_in[18];
  const float* be1    = (const float*)d_in[19];
  const float* g2     = (const float*)d_in[20];
  const float* be2    = (const float*)d_in[21];
  const float* g3     = (const float*)d_in[22];
  const float* be3    = (const float*)d_in[23];
  float* out = (float*)d_out;

  // ---- workspace layout
  float* ws   = (float*)d_ws;
  float* qkv  = ws;                              // 4096*3072 f32
  float* attn = qkv  + (size_t)MTOK * 3 * D_;    // 4096*1024 f32
  float* proj = attn + (size_t)MTOK * D_;
  float* h1   = proj + (size_t)MTOK * D_;
  float* ffn1 = h1   + (size_t)MTOK * D_;        // 4096*4096 f32
  float* qca  = qkv;                             // reuse qkv region
  float* kca  = qkv + (size_t)MTOK * D_;
  float* vca  = qkv + (size_t)2 * MTOK * D_;
  float* h2   = attn;                            // reuse after cross-proj

  __bf16* bws    = (__bf16*)(ffn1 + (size_t)MTOK * DFF_);
  __bf16* wqkvT  = bws;                          // [3072][1024]
  __bf16* wsaoT  = wqkvT + (size_t)3 * D_ * D_;  // [1024][1024]
  __bf16* wqT    = wsaoT + (size_t)D_ * D_;
  __bf16* wkT    = wqT   + (size_t)D_ * D_;
  __bf16* wvT    = wkT   + (size_t)D_ * D_;
  __bf16* wcaoT  = wvT   + (size_t)D_ * D_;
  __bf16* w1T    = wcaoT + (size_t)D_ * D_;      // [4096][1024]
  __bf16* w2T    = w1T   + (size_t)D_ * DFF_;    // [1024][4096]
  __bf16* abf    = w2T   + (size_t)DFF_ * D_;    // up to 4096*4096 bf16

  const dim3 blk(256);
  const dim3 ablk(128);
  const dim3 agrid(S_ / 64, H_, B_);
  auto ggrid = [](int M, int N) { return dim3(N / 256, M / 64); };
  auto tgrid = [](int Kd, int Nd) { return dim3(Nd / 32, Kd / 32); };
  auto cgrid = [](size_t n) { return dim3((unsigned)(n / 1024)); };

  // ---- one-time weight packs (f32 [K][N] -> bf16 [N][K])
  transpose_bf16_kernel<<<tgrid(D_, 3 * D_), blk, 0, stream>>>(w_qkv, wqkvT, D_, 3 * D_);
  transpose_bf16_kernel<<<tgrid(D_, D_), blk, 0, stream>>>(w_sa_o, wsaoT, D_, D_);
  transpose_bf16_kernel<<<tgrid(D_, D_), blk, 0, stream>>>(w_q, wqT, D_, D_);
  transpose_bf16_kernel<<<tgrid(D_, D_), blk, 0, stream>>>(w_k, wkT, D_, D_);
  transpose_bf16_kernel<<<tgrid(D_, D_), blk, 0, stream>>>(w_v, wvT, D_, D_);
  transpose_bf16_kernel<<<tgrid(D_, D_), blk, 0, stream>>>(w_ca_o, wcaoT, D_, D_);
  transpose_bf16_kernel<<<tgrid(D_, DFF_), blk, 0, stream>>>(w1, w1T, D_, DFF_);
  transpose_bf16_kernel<<<tgrid(DFF_, D_), blk, 0, stream>>>(w2, w2T, DFF_, D_);

  // 1. fused qkv projection
  cvt_bf16_kernel<<<cgrid((size_t)MTOK * D_), blk, 0, stream>>>(x, abf);
  gemm_tdm_kernel<0><<<ggrid(MTOK, 3 * D_), blk, 0, stream>>>(
      abf, wqkvT, b_qkv, qkv, MTOK, 3 * D_, D_);
  // 2. masked self-attention (interleaved head layout via strides)
  attn_kernel<<<agrid, ablk, 0, stream>>>(
      qkv, qkv + DH_, qkv + 2 * DH_, attn,
      (long)S_ * 3 * D_, 3 * D_, 3 * DH_,
      (long)S_ * 3 * D_, 3 * D_, 3 * DH_,
      (long)S_ * 3 * D_, 3 * D_, 3 * DH_,
      S_, 1);
  // 3. self-attn output projection
  cvt_bf16_kernel<<<cgrid((size_t)MTOK * D_), blk, 0, stream>>>(attn, abf);
  gemm_tdm_kernel<0><<<ggrid(MTOK, D_), blk, 0, stream>>>(
      abf, wsaoT, b_sa_o, proj, MTOK, D_, D_);
  // 4. h1 = LN(x + proj)
  add_ln_kernel<<<dim3(MTOK), blk, 0, stream>>>(x, proj, g1, be1, h1);
  // 5-7. cross-attention projections
  cvt_bf16_kernel<<<cgrid((size_t)MTOK * D_), blk, 0, stream>>>(h1, abf);
  gemm_tdm_kernel<0><<<ggrid(MTOK, D_), blk, 0, stream>>>(
      abf, wqT, b_q, qca, MTOK, D_, D_);
  cvt_bf16_kernel<<<cgrid((size_t)MTOK * D_), blk, 0, stream>>>(enc_x, abf);
  gemm_tdm_kernel<0><<<ggrid(MTOK, D_), blk, 0, stream>>>(
      abf, wkT, b_k, kca, MTOK, D_, D_);
  gemm_tdm_kernel<0><<<ggrid(MTOK, D_), blk, 0, stream>>>(
      abf, wvT, b_v, vca, MTOK, D_, D_);
  // 8. cross attention
  attn_kernel<<<agrid, ablk, 0, stream>>>(
      qca, kca, vca, attn,
      (long)S_ * D_, D_, DH_,
      (long)E_ * D_, D_, DH_,
      (long)E_ * D_, D_, DH_,
      E_, 0);
  // 9. cross-attn output projection
  cvt_bf16_kernel<<<cgrid((size_t)MTOK * D_), blk, 0, stream>>>(attn, abf);
  gemm_tdm_kernel<0><<<ggrid(MTOK, D_), blk, 0, stream>>>(
      abf, wcaoT, b_ca_o, proj, MTOK, D_, D_);
  // 10. h2 = LN(h1 + proj)
  add_ln_kernel<<<dim3(MTOK), blk, 0, stream>>>(h1, proj, g2, be2, h2);
  // 11. ffn1 = gelu(h2 @ w1 + b1)
  cvt_bf16_kernel<<<cgrid((size_t)MTOK * D_), blk, 0, stream>>>(h2, abf);
  gemm_tdm_kernel<1><<<ggrid(MTOK, DFF_), blk, 0, stream>>>(
      abf, w1T, b1, ffn1, MTOK, DFF_, D_);
  // 12. ffn2 = ffn1 @ w2 + b2
  cvt_bf16_kernel<<<cgrid((size_t)MTOK * DFF_), blk, 0, stream>>>(ffn1, abf);
  gemm_tdm_kernel<0><<<ggrid(MTOK, D_), blk, 0, stream>>>(
      abf, w2T, b2, proj, MTOK, D_, DFF_);
  // 13. out = LN(h2 + ffn2)
  add_ln_kernel<<<dim3(MTOK), blk, 0, stream>>>(h2, proj, g3, be3, out);
}
